// PointNetFeaturePropagation_9242769621827
// MI455X (gfx1250) — matrix operand order repaired and checked by reading, not compiled, not run on hardware
//
#include <hip/hip_runtime.h>

typedef __bf16 v16bf __attribute__((ext_vector_type(16)));
typedef float  v8f   __attribute__((ext_vector_type(8)));

#define BN   8
#define NPTS 8192
#define SPTS 2048
#define D1   128
#define D2   256
#define CIN  384   // D1 + D2
#define CMID 256
#define COUT 128

__device__ __forceinline__ unsigned short f2bf(float f) {
  unsigned u = __float_as_uint(f);
  u += 0x7FFFu + ((u >> 16) & 1u);   // round-to-nearest-even into bf16
  return (unsigned short)(u >> 16);
}

// ---------------------------------------------------------------------------
// Pack W [M][K] f32 row-major -> per-lane WMMA A-fragments, bf16.
// Layout: Apack[mo][kstep][lane][e], e=0..15, one contiguous 32B per lane.
// Contiguous K order (matches B): lane<16 holds K 0-15 of row M=lane,
// lane>=16 holds K 16-31 of row M=lane-16, 2 bf16 per VGPR.
// ---------------------------------------------------------------------------
template <int M, int K>
__global__ __launch_bounds__(256) void pack_weights(const float* __restrict__ W,
                                                    unsigned short* __restrict__ Apack) {
  constexpr int KSTEPS = K / 32;
  int idx = blockIdx.x * 256 + threadIdx.x;
  if (idx >= M * K) return;
  int e     = idx & 15;
  int lane  = (idx >> 4) & 31;
  int kstep = (idx >> 9) % KSTEPS;
  int mo    = (idx >> 9) / KSTEPS;
  int lhalf = lane >> 4;
  int row   = mo * 16 + (lane & 15);
  int klocal = e + lhalf * 16;
  Apack[idx] = f2bf(W[row * K + kstep * 32 + klocal]);
}

// ---------------------------------------------------------------------------
// kNN-3 + inverse-distance interpolation + concat, emitting bf16 [b][n][384].
// One block = 256 query points; xyz2 tile cached in LDS.
// ---------------------------------------------------------------------------
__global__ __launch_bounds__(256) void knn_interp(const float* __restrict__ xyz1,
                                                  const float* __restrict__ xyz2,
                                                  const float* __restrict__ points1,
                                                  const float* __restrict__ points2,
                                                  unsigned short* __restrict__ newp) {
  __shared__ float sx[SPTS], sy[SPTS], sz[SPTS];
  __shared__ int   sIdx[256 * 3];
  __shared__ float sWt[256 * 3];
  int b   = blockIdx.x / (NPTS / 256);
  int n0  = (blockIdx.x % (NPTS / 256)) * 256;
  int tid = threadIdx.x;

  const float* x2 = xyz2 + (size_t)b * SPTS * 3;
  for (int i = tid; i < SPTS; i += 256) {
    sx[i] = x2[i * 3 + 0];
    sy[i] = x2[i * 3 + 1];
    sz[i] = x2[i * 3 + 2];
  }
  __syncthreads();

  int n = n0 + tid;
  const float* p = xyz1 + ((size_t)b * NPTS + n) * 3;
  float px = p[0], py = p[1], pz = p[2];
  float d0 = 3.4e38f, d1 = 3.4e38f, d2 = 3.4e38f;
  int   i0 = 0, i1 = 0, i2 = 0;
  for (int s = 0; s < SPTS; ++s) {
    float dx = px - sx[s], dy = py - sy[s], dz = pz - sz[s];
    float d  = dx * dx + dy * dy + dz * dz;
    if (d < d2) {
      if (d < d1) {
        d2 = d1; i2 = i1;
        if (d < d0) { d1 = d0; i1 = i0; d0 = d; i0 = s; }
        else        { d1 = d;  i1 = s; }
      } else { d2 = d; i2 = s; }
    }
  }
  float r0 = 1.f / (d0 + 1e-8f), r1 = 1.f / (d1 + 1e-8f), r2 = 1.f / (d2 + 1e-8f);
  float rs = 1.f / (r0 + r1 + r2);
  sIdx[tid * 3 + 0] = i0;      sIdx[tid * 3 + 1] = i1;      sIdx[tid * 3 + 2] = i2;
  sWt[tid * 3 + 0] = r0 * rs;  sWt[tid * 3 + 1] = r1 * rs;  sWt[tid * 3 + 2] = r2 * rs;
  __syncthreads();

  // interpolated channels -> c in [128, 384), one channel per thread per point
  const float* p2 = points2 + (size_t)b * SPTS * D2;
  for (int pt = 0; pt < 256; ++pt) {
    int   j0 = sIdx[pt * 3 + 0], j1 = sIdx[pt * 3 + 1], j2 = sIdx[pt * 3 + 2];
    float w0 = sWt[pt * 3 + 0],  w1 = sWt[pt * 3 + 1],  w2 = sWt[pt * 3 + 2];
    float v = w0 * p2[(size_t)j0 * D2 + tid] +
              w1 * p2[(size_t)j1 * D2 + tid] +
              w2 * p2[(size_t)j2 * D2 + tid];
    newp[((size_t)b * NPTS + n0 + pt) * CIN + D1 + tid] = f2bf(v);
  }
  // copy points1 -> channels [0, 128)
  const float* p1 = points1 + (size_t)b * NPTS * D1;
  for (int k = 0; k < 128; ++k) {
    int flat = k * 256 + tid;
    int pn   = flat / D1;
    int c    = flat % D1;
    newp[((size_t)b * NPTS + n0 + pn) * CIN + c] =
        f2bf(p1[(size_t)(n0 + pn) * D1 + c]);
  }
}

// ---------------------------------------------------------------------------
// Y[b][o][n] = sum_c W[o][c] * X[b][n][c] via V_WMMA_F32_16X16X32_BF16.
// Block = 8 waves covering full M and 64 columns; wave = MT 16-row tiles x 4
// 16-col tiles. Every A/B fragment is a single 32B contiguous load.
// ---------------------------------------------------------------------------
template <int K, int M, int MT>
__global__ __launch_bounds__(256) void gemm_wmma(const unsigned short* __restrict__ Apack,
                                                 const unsigned short* __restrict__ X,
                                                 float* __restrict__ Y) {
  constexpr int KSTEPS = K / 32;
  int b     = blockIdx.y;
  int n0    = blockIdx.x * 64;
  int wave  = threadIdx.x >> 5;
  int lane  = threadIdx.x & 31;
  int lhalf = lane >> 4;
  int l16   = lane & 15;

  const v8f vzero = {0.f, 0.f, 0.f, 0.f, 0.f, 0.f, 0.f, 0.f};
  v8f acc[MT][4];
#pragma unroll
  for (int m = 0; m < MT; ++m)
#pragma unroll
    for (int j = 0; j < 4; ++j) acc[m][j] = vzero;

  const unsigned short* Xb = X + (size_t)b * NPTS * K;
  const unsigned short* brow[4];
#pragma unroll
  for (int j = 0; j < 4; ++j)
    brow[j] = Xb + (size_t)(n0 + j * 16 + l16) * K + lhalf * 16;
  const unsigned short* arow = Apack + ((size_t)(wave * MT) * KSTEPS * 32 + lane) * 16;

  for (int ks = 0; ks < KSTEPS; ++ks) {
    v16bf bf[4];
#pragma unroll
    for (int j = 0; j < 4; ++j)
      bf[j] = *(const v16bf*)(brow[j] + ks * 32);
#pragma unroll
    for (int m = 0; m < MT; ++m) {
      v16bf af = *(const v16bf*)(arow + (size_t)(m * KSTEPS + ks) * 32 * 16);
#pragma unroll
      for (int j = 0; j < 4; ++j)
        acc[m][j] = __builtin_amdgcn_wmma_f32_16x16x32_bf16(
            false, af, false, bf[j], (short)0, acc[m][j], false, false);
    }
  }

  // C/D layout (ISA 7.12.2): VGPR r holds row r+lhalf*8, col = lane%16.
#pragma unroll
  for (int m = 0; m < MT; ++m) {
    int obase = (wave * MT + m) * 16 + lhalf * 8;
#pragma unroll
    for (int j = 0; j < 4; ++j) {
      int col = n0 + j * 16 + l16;
#pragma unroll
      for (int r = 0; r < 8; ++r)
        Y[((size_t)b * M + obase + r) * NPTS + col] = acc[m][j][r];
    }
  }
}

// ---------------------------------------------------------------------------
// Per-(b,o) mean / rstd over N points (biased variance, eps = 1e-5).
// ---------------------------------------------------------------------------
__global__ __launch_bounds__(256) void row_stats(const float* __restrict__ Y,
                                                 float* __restrict__ stats) {
  __shared__ float ssum[256], ssq[256];
  const float* row = Y + (size_t)blockIdx.x * NPTS;
  float s = 0.f, q = 0.f;
  for (int i = threadIdx.x; i < NPTS; i += 256) {
    float v = row[i];
    s += v;
    q += v * v;
  }
  ssum[threadIdx.x] = s;
  ssq[threadIdx.x]  = q;
  __syncthreads();
  for (int off = 128; off > 0; off >>= 1) {
    if (threadIdx.x < off) {
      ssum[threadIdx.x] += ssum[threadIdx.x + off];
      ssq[threadIdx.x]  += ssq[threadIdx.x + off];
    }
    __syncthreads();
  }
  if (threadIdx.x == 0) {
    float mean = ssum[0] * (1.f / NPTS);
    float var  = ssq[0] * (1.f / NPTS) - mean * mean;
    stats[blockIdx.x * 2 + 0] = mean;
    stats[blockIdx.x * 2 + 1] = rsqrtf(var + 1e-5f);
  }
}

// ---------------------------------------------------------------------------
// Normalize + ReLU + transpose [b][o][n] -> [b][n][C] via 32x32 LDS tile.
// BF16OUT=true emits bf16 rows (layer-2 B fragments), else f32 to d_out.
// ---------------------------------------------------------------------------
template <int C, bool BF16OUT>
__global__ __launch_bounds__(256) void norm_transpose(const float* __restrict__ Y,
                                                      const float* __restrict__ stats,
                                                      void* __restrict__ outv) {
  __shared__ float tile[32][33];
  int b  = blockIdx.z;
  int o0 = blockIdx.x * 32;
  int n0 = blockIdx.y * 32;
  int tx = threadIdx.x, ty = threadIdx.y;   // 32 x 8
#pragma unroll
  for (int i = 0; i < 4; ++i) {
    int   o    = o0 + ty + i * 8;
    float mean = stats[(b * C + o) * 2 + 0];
    float rstd = stats[(b * C + o) * 2 + 1];
    float v    = Y[((size_t)b * C + o) * NPTS + n0 + tx];
    v = (v - mean) * rstd;
    tile[ty + i * 8][tx] = v > 0.f ? v : 0.f;
  }
  __syncthreads();
#pragma unroll
  for (int i = 0; i < 4; ++i) {
    int n = n0 + ty + i * 8;
    int o = o0 + tx;
    if (BF16OUT)
      ((unsigned short*)outv)[((size_t)b * NPTS + n) * C + o] = f2bf(tile[tx][ty + i * 8]);
    else
      ((float*)outv)[((size_t)b * NPTS + n) * C + o] = tile[tx][ty + i * 8];
  }
}

// ---------------------------------------------------------------------------
extern "C" void kernel_launch(void* const* d_in, const int* in_sizes, int n_in,
                              void* d_out, int out_size, void* d_ws, size_t ws_size,
                              hipStream_t stream) {
  (void)in_sizes; (void)n_in; (void)out_size; (void)ws_size;
  const float* xyz1    = (const float*)d_in[0];
  const float* xyz2    = (const float*)d_in[1];
  const float* points1 = (const float*)d_in[2];
  const float* points2 = (const float*)d_in[3];
  const float* W1      = (const float*)d_in[4];
  // d_in[5] = b1, d_in[7] = b2: biases cancel exactly under InstanceNorm.
  const float* W2      = (const float*)d_in[6];

  char* ws = (char*)d_ws;
  // Workspace layout (aliased):
  //   XA : bf16 activations — newp [b][n][384] for layer 1, then reused as
  //        X2 [b][n][256] for layer 2 (48 MB region)
  //   YA : f32 pre-norm GEMM output, Y1 then Y2 (64 MB region)
  const size_t OFS_XA  = 0;
  const size_t OFS_AP1 = (size_t)BN * NPTS * CIN * 2;        // 50,331,648
  const size_t OFS_AP2 = OFS_AP1 + (size_t)CMID * CIN * 2;   // + 196,608
  const size_t OFS_YA  = OFS_AP2 + (size_t)COUT * CMID * 2;  // + 65,536
  const size_t OFS_ST1 = OFS_YA + (size_t)BN * CMID * NPTS * 4;
  const size_t OFS_ST2 = OFS_ST1 + (size_t)BN * CMID * 2 * 4;

  unsigned short* XA  = (unsigned short*)(ws + OFS_XA);
  unsigned short* Ap1 = (unsigned short*)(ws + OFS_AP1);
  unsigned short* Ap2 = (unsigned short*)(ws + OFS_AP2);
  float*          YA  = (float*)(ws + OFS_YA);
  float*          st1 = (float*)(ws + OFS_ST1);
  float*          st2 = (float*)(ws + OFS_ST2);

  // 1) weight packing into WMMA A-fragment layout
  pack_weights<CMID, CIN><<<(CMID * CIN + 255) / 256, 256, 0, stream>>>(W1, Ap1);
  pack_weights<COUT, CMID><<<(COUT * CMID + 255) / 256, 256, 0, stream>>>(W2, Ap2);

  // 2) kNN-3 + interpolation + concat -> bf16 [b][n][384]
  knn_interp<<<BN * (NPTS / 256), 256, 0, stream>>>(xyz1, xyz2, points1, points2, XA);

  // 3) layer 1 GEMM (M=256, K=384) -> Y1 f32 [b][o][n]
  gemm_wmma<CIN, CMID, 2><<<dim3(NPTS / 64, BN), 256, 0, stream>>>(Ap1, XA, YA);
  row_stats<<<BN * CMID, 256, 0, stream>>>(YA, st1);
  norm_transpose<CMID, true><<<dim3(CMID / 32, NPTS / 32, BN), dim3(32, 8), 0, stream>>>(YA, st1, XA);

  // 4) layer 2 GEMM (M=128, K=256) -> Y2 f32 [b][o][n] (reuses YA)
  gemm_wmma<CMID, COUT, 1><<<dim3(NPTS / 64, BN), 256, 0, stream>>>(Ap2, XA, YA);
  row_stats<<<BN * COUT, 256, 0, stream>>>(YA, st2);
  norm_transpose<COUT, false><<<dim3(COUT / 32, NPTS / 32, BN), dim3(32, 8), 0, stream>>>(YA, st2, d_out);
}
